// MT5Generation_50225347560136
// MI455X (gfx1250) — matrix operand order repaired and checked by reading, not compiled, not run on hardware
//
#include <hip/hip_runtime.h>

// ---- problem constants (match reference) ----
#define Bn  4
#define SQn 1024
#define SKn 1024
#define Dm  1024
#define Hh  16
#define DHd 64
#define FFn 4096
#define NBn 32
#define EPSc 1e-6f

typedef __attribute__((ext_vector_type(16))) __bf16 v16bf;
typedef __attribute__((ext_vector_type(8)))  float  v8f;

union FragBF { uint4 u[2]; v16bf v; };

// ---- async global->LDS copy (CDNA5 ASYNCcnt path), guarded fallback ----
#if defined(__has_builtin)
#  if __has_builtin(__builtin_amdgcn_global_load_async_to_lds_b128) && \
      __has_builtin(__builtin_amdgcn_s_wait_asynccnt)
#    define USE_ASYNC_COPY 1
#  endif
#endif
#ifndef USE_ASYNC_COPY
#  define USE_ASYNC_COPY 0
#endif

#if USE_ASYNC_COPY
typedef int v4i_vs __attribute__((vector_size(16)));                 // matches builtin pointee
typedef __attribute__((address_space(1))) v4i_vs* gv4i_p;            // global (AS1)
typedef __attribute__((address_space(3))) v4i_vs* lv4i_p;            // LDS (AS3)
#endif

__device__ __forceinline__ void cp16_async(const unsigned short* g, unsigned short* l) {
#if USE_ASYNC_COPY
  __builtin_amdgcn_global_load_async_to_lds_b128((gv4i_p)g, (lv4i_p)l, 0, 0);
#else
  *(uint4*)l = *(const uint4*)g;
#endif
}
__device__ __forceinline__ void cp_commit_wait() {
#if USE_ASYNC_COPY
  __builtin_amdgcn_s_wait_asynccnt(0);
#endif
}

__device__ __forceinline__ unsigned short f2bf(float f) {
  unsigned u = __float_as_uint(f);
  u += 0x7FFFu + ((u >> 16) & 1u);           // round-to-nearest-even
  return (unsigned short)(u >> 16);
}

__device__ __forceinline__ float gelu_new_f(float x) {
  return 0.5f * x * (1.0f + tanhf(0.7978845608028654f * (x + 0.044715f * x * x * x)));
}

__device__ __forceinline__ v8f wmma_bf16(v16bf a, v16bf b, v8f c) {
  // D = A(16x32 bf16) * B(32x16 bf16) + C(16x16 f32)
  return __builtin_amdgcn_wmma_f32_16x16x32_bf16(false, a, false, b, (short)0, c, false, false);
}

__device__ __forceinline__ float redmax16(float v) {
  v = fmaxf(v, __shfl_xor(v, 1, 32));
  v = fmaxf(v, __shfl_xor(v, 2, 32));
  v = fmaxf(v, __shfl_xor(v, 4, 32));
  v = fmaxf(v, __shfl_xor(v, 8, 32));
  return v;
}
__device__ __forceinline__ float redsum16(float v) {
  v += __shfl_xor(v, 1, 32);
  v += __shfl_xor(v, 2, 32);
  v += __shfl_xor(v, 4, 32);
  v += __shfl_xor(v, 8, 32);
  return v;
}

// ---------------- f32 -> bf16 convert (activations) ----------------
__global__ void cvt_f32_bf16_kernel(const float* __restrict__ src,
                                    unsigned short* __restrict__ dst, int n) {
  int i = blockIdx.x * blockDim.x + threadIdx.x;
  int stride = gridDim.x * blockDim.x;
  for (; i < n; i += stride) dst[i] = f2bf(src[i]);
}

// ---------------- f32 [K][N] -> bf16 [N][K] (weights, one-time) ----------------
__global__ __launch_bounds__(256)
void cvt_transpose_bf16_kernel(const float* __restrict__ src,
                               unsigned short* __restrict__ dst, int K, int N) {
  __shared__ unsigned short st[32][33];
  const int n0 = blockIdx.x * 32;
  const int k0 = blockIdx.y * 32;
  const int c = threadIdx.x & 31;
  const int r = threadIdx.x >> 5;   // 0..7
#pragma unroll
  for (int i = 0; i < 4; ++i) {
    const int k = r + i * 8;
    st[k][c] = f2bf(src[(size_t)(k0 + k) * N + n0 + c]);
  }
  __syncthreads();
#pragma unroll
  for (int i = 0; i < 4; ++i) {
    const int n = r + i * 8;
    dst[(size_t)(n0 + n) * K + k0 + c] = st[c][n];
  }
}

// ---------------- T5 RMSNorm -> bf16 ----------------
__global__ __launch_bounds__(256)
void rmsnorm_bf16_kernel(const float* __restrict__ x, const float* __restrict__ w,
                         unsigned short* __restrict__ out) {
  __shared__ float red[8];
  const int row = blockIdx.x;
  const float4 xv = ((const float4*)(x + (size_t)row * Dm))[threadIdx.x];
  float ss = xv.x * xv.x + xv.y * xv.y + xv.z * xv.z + xv.w * xv.w;
  ss += __shfl_xor(ss, 16, 32);
  ss += __shfl_xor(ss, 8, 32);
  ss += __shfl_xor(ss, 4, 32);
  ss += __shfl_xor(ss, 2, 32);
  ss += __shfl_xor(ss, 1, 32);
  if ((threadIdx.x & 31) == 0) red[threadIdx.x >> 5] = ss;
  __syncthreads();
  const float tot = red[0] + red[1] + red[2] + red[3] + red[4] + red[5] + red[6] + red[7];
  const float rs = rsqrtf(tot * (1.0f / Dm) + EPSc);
  const float4 wv = ((const float4*)w)[threadIdx.x];
  unsigned short* o = out + (size_t)row * Dm + threadIdx.x * 4;
  o[0] = f2bf(xv.x * rs * wv.x);
  o[1] = f2bf(xv.y * rs * wv.y);
  o[2] = f2bf(xv.z * rs * wv.z);
  o[3] = f2bf(xv.w * rs * wv.w);
}

// ---------------- bf16 WMMA GEMM: C[M,N] = A[M,K] x B^T[N,K] ----------------
// B is pre-transposed [N][K] so every tile stages as contiguous 16B async copies.
// EPI 0: f32 | 1: f32 + resid | 2: bf16 | 3: gated bf16(gelu(A@B0)*(A@B1)) | 4: bf16 V^T scatter
#define BM 128
#define BN 64
#define BK 32
#define APAD 40   // ushort row stride (80B: 16B aligned + bank-conflict padding)
#define BPAD 40

template <int EPI>
__global__ __launch_bounds__(256)
void gemm_bf16_kernel(const unsigned short* __restrict__ A,
                      const unsigned short* __restrict__ B0,
                      const unsigned short* __restrict__ B1,
                      const float* __restrict__ resid,
                      float* __restrict__ Cf,
                      unsigned short* __restrict__ Cbf,
                      int M, int N, int K) {
  __shared__ unsigned short sA[2][BM][APAD];
  __shared__ unsigned short sB0[2][BN][BPAD];
  __shared__ unsigned short sB1[(EPI == 3) ? 2 : 1][BN][BPAD];

  const int tid = threadIdx.x;
  const int wid = tid >> 5;
  const int lane = tid & 31;
  const int wm = wid & 3;            // 4 waves along M
  const int wn = wid >> 2;           // 2 waves along N
  const int tileM = blockIdx.y * BM;
  const int tileN = blockIdx.x * BN;

  v8f acc[2][2];
  v8f accB[2][2];
#pragma unroll
  for (int i = 0; i < 2; ++i)
#pragma unroll
    for (int j = 0; j < 2; ++j) { acc[i][j] = {}; accB[i][j] = {}; }

  // staging coordinates
  const int arow = tid >> 1;          // 0..127
  const int acol = (tid & 1) * 16;    // 0 / 16
  const int brow = tid >> 2;          // 0..63 (N index of B^T)
  const int bkc  = (tid & 3) * 8;     // 0..24 (K chunk)

  // fragment coordinates (contiguous k order: k = 16*(lane/16) + s for A and B)
  const int am = lane & 15;
  const int akb = (lane >> 4) * 16;
  const int bn = lane & 15;
  const int bkb = (lane >> 4) * 16;

  auto stage = [&](int p, int k0) {
    const unsigned short* ag = A + (size_t)(tileM + arow) * K + k0 + acol;
    cp16_async(ag, &sA[p][arow][acol]);
    cp16_async(ag + 8, &sA[p][arow][acol + 8]);
    cp16_async(B0 + (size_t)(tileN + brow) * K + k0 + bkc, &sB0[p][brow][bkc]);
    if (EPI == 3)
      cp16_async(B1 + (size_t)(tileN + brow) * K + k0 + bkc, &sB1[p][brow][bkc]);
  };

  stage(0, 0);
  cp_commit_wait();
  __syncthreads();

  int p = 0;
  for (int k0 = 0; k0 < K; k0 += BK, p ^= 1) {
    const bool more = (k0 + BK) < K;
    if (more) stage(p ^ 1, k0 + BK);   // overlap next-tile DMA with compute

    FragBF afr[2], bfr[2], bfr1[2];
#pragma unroll
    for (int i = 0; i < 2; ++i) {
      const int row = wm * 32 + i * 16 + am;
      afr[i].u[0] = *(const uint4*)&sA[p][row][akb];
      afr[i].u[1] = *(const uint4*)&sA[p][row][akb + 8];
    }
#pragma unroll
    for (int j = 0; j < 2; ++j) {
      const int col = wn * 32 + j * 16 + bn;
      bfr[j].u[0] = *(const uint4*)&sB0[p][col][bkb];
      bfr[j].u[1] = *(const uint4*)&sB0[p][col][bkb + 8];
      if (EPI == 3) {
        bfr1[j].u[0] = *(const uint4*)&sB1[p][col][bkb];
        bfr1[j].u[1] = *(const uint4*)&sB1[p][col][bkb + 8];
      }
    }
#pragma unroll
    for (int i = 0; i < 2; ++i)
#pragma unroll
      for (int j = 0; j < 2; ++j) {
        acc[i][j] = wmma_bf16(afr[i].v, bfr[j].v, acc[i][j]);
        if (EPI == 3) accB[i][j] = wmma_bf16(afr[i].v, bfr1[j].v, accB[i][j]);
      }

    if (more) {
      cp_commit_wait();
      __syncthreads();
    }
  }

  // ---- epilogue (C layout: lane%16 = N col; rows r + 8*(lane>=16)) ----
  const int mhal = (lane >> 4) * 8;
#pragma unroll
  for (int i = 0; i < 2; ++i) {
    const int mg = tileM + wm * 32 + i * 16 + mhal;
#pragma unroll
    for (int j = 0; j < 2; ++j) {
      const int ng = tileN + wn * 32 + j * 16 + (lane & 15);
#pragma unroll
      for (int r = 0; r < 8; ++r) {
        const int row = mg + r;
        const float v = acc[i][j][r];
        if (EPI == 0) {
          Cf[(size_t)row * N + ng] = v;
        } else if (EPI == 1) {
          const size_t idx = (size_t)row * N + ng;
          Cf[idx] = resid[idx] + v;
        } else if (EPI == 2) {
          Cbf[(size_t)row * N + ng] = f2bf(v);
        } else if (EPI == 3) {
          Cbf[(size_t)row * N + ng] = f2bf(gelu_new_f(v) * accB[i][j][r]);
        } else {
          // V^T layout [b][h][d][key]: row = b*SK + key, ng = h*64 + d
          const int bb = row >> 10, key = row & (SKn - 1);
          const int hh = ng >> 6, dd = ng & (DHd - 1);
          Cbf[(((size_t)(bb * Hh + hh) * DHd + dd) << 10) + key] = f2bf(v);
        }
      }
    }
  }
}

// ---------------- fused flash attention with T5 relative bias ----------------
// grid: (SQ/64, H, B), block: 128 (4 waves; wave owns 16 query rows)
// Vt is pre-transposed: [b][h][d][key]
__global__ __launch_bounds__(128)
void flash_attn_kernel(const unsigned short* __restrict__ Qm,
                       const unsigned short* __restrict__ Km,
                       const unsigned short* __restrict__ Vt,
                       const float* __restrict__ rel_bias,
                       const int* __restrict__ buckets,
                       unsigned short* __restrict__ Ctx) {
  __shared__ unsigned short sK[64][72];     // [key][dh]
  __shared__ unsigned short sVT[64][72];    // [dh][key]
  __shared__ unsigned short sP[4][16][72];  // per-wave P (C-layout -> A-layout bounce)
  __shared__ float sBias[NBn * Hh];

  const int tid = threadIdx.x;
  const int wid = tid >> 5;
  const int lane = tid & 31;
  const int halo = lane >> 4;   // 0/1
  const int ln = lane & 15;
  const int b = blockIdx.z, h = blockIdx.y;
  const int q0 = blockIdx.x * 64;

  for (int i = tid; i < NBn * Hh; i += 128) sBias[i] = rel_bias[i];
  __syncthreads();

  // Q fragments for this wave's 16 rows (dh = 0..63), in registers for whole kernel
  FragBF qf[2];
  {
    const unsigned short* qrow =
        Qm + ((size_t)(b * SQn + q0 + wid * 16 + ln)) * (Hh * DHd) + h * DHd;
    const int kb = halo * 16;
    qf[0].u[0] = *(const uint4*)(qrow + kb);
    qf[0].u[1] = *(const uint4*)(qrow + kb + 8);
    qf[1].u[0] = *(const uint4*)(qrow + 32 + kb);
    qf[1].u[1] = *(const uint4*)(qrow + 32 + kb + 8);
  }

  v8f O[4];
#pragma unroll
  for (int nt = 0; nt < 4; ++nt) O[nt] = {};
  float mrun[8], lrun[8];
#pragma unroll
  for (int r = 0; r < 8; ++r) { mrun[r] = -1e30f; lrun[r] = 0.0f; }

  // staging coordinates (128 threads cover 64x64 tiles, 32 elem each)
  const int srow = tid >> 1;
  const int scb = (tid & 1) * 32;

  for (int kt = 0; kt < SKn; kt += 64) {
    {  // K tile rows and V^T tile rows are both contiguous -> pure async B128 copies
      const unsigned short* kg =
          Km + ((size_t)(b * SKn + kt + srow)) * (Hh * DHd) + h * DHd + scb;
      cp16_async(kg,      &sK[srow][scb]);
      cp16_async(kg + 8,  &sK[srow][scb + 8]);
      cp16_async(kg + 16, &sK[srow][scb + 16]);
      cp16_async(kg + 24, &sK[srow][scb + 24]);
      const unsigned short* vg =
          Vt + ((size_t)((b * Hh + h) * DHd + srow)) * SKn + kt + scb;
      cp16_async(vg,      &sVT[srow][scb]);
      cp16_async(vg + 8,  &sVT[srow][scb + 8]);
      cp16_async(vg + 16, &sVT[srow][scb + 16]);
      cp16_async(vg + 24, &sVT[srow][scb + 24]);
    }
    cp_commit_wait();
    __syncthreads();

    // ---- S = Q * K^T  (16 x 64 strip per wave) ----
    v8f S[4];
#pragma unroll
    for (int nt = 0; nt < 4; ++nt) {
      FragBF kf0, kf1;
      const int col = nt * 16 + ln;       // key column
      const int kb2 = halo * 16;          // dh chunk (contiguous k = 16*half + s)
      kf0.u[0] = *(const uint4*)&sK[col][kb2];
      kf0.u[1] = *(const uint4*)&sK[col][kb2 + 8];
      kf1.u[0] = *(const uint4*)&sK[col][32 + kb2];
      kf1.u[1] = *(const uint4*)&sK[col][32 + kb2 + 8];
      v8f s = {};
      s = wmma_bf16(qf[0].v, kf0.v, s);
      s = wmma_bf16(qf[1].v, kf1.v, s);
      S[nt] = s;
    }

    // ---- add relative-position bias + online softmax ----
    float alpha[8];
#pragma unroll
    for (int r = 0; r < 8; ++r) {
      const int qidx = q0 + wid * 16 + r + 8 * halo;
#pragma unroll
      for (int nt = 0; nt < 4; ++nt) {
        const int key = kt + nt * 16 + ln;
        const int bk = buckets[(size_t)qidx * SKn + key];
        S[nt][r] = S[nt][r] + sBias[bk * Hh + h];
      }
      float mx = fmaxf(fmaxf(S[0][r], S[1][r]), fmaxf(S[2][r], S[3][r]));
      mx = redmax16(mx);
      const float mnew = fmaxf(mrun[r], mx);
      alpha[r] = __expf(mrun[r] - mnew);
      float rsum = 0.0f;
#pragma unroll
      for (int nt = 0; nt < 4; ++nt) {
        const float pv = __expf(S[nt][r] - mnew);
        S[nt][r] = pv;
        rsum += pv;
      }
      rsum = redsum16(rsum);
      lrun[r] = lrun[r] * alpha[r] + rsum;
      mrun[r] = mnew;
    }
#pragma unroll
    for (int nt = 0; nt < 4; ++nt)
#pragma unroll
      for (int r = 0; r < 8; ++r) O[nt][r] = O[nt][r] * alpha[r];

    // ---- P: C layout -> A layout via per-wave LDS bounce ----
#pragma unroll
    for (int nt = 0; nt < 4; ++nt)
#pragma unroll
      for (int r = 0; r < 8; ++r)
        sP[wid][r + 8 * halo][nt * 16 + ln] = f2bf(S[nt][r]);

    FragBF pf[2];
    {
      const int kb = halo * 16;
      pf[0].u[0] = *(const uint4*)&sP[wid][ln][kb];
      pf[0].u[1] = *(const uint4*)&sP[wid][ln][kb + 8];
      pf[1].u[0] = *(const uint4*)&sP[wid][ln][32 + kb];
      pf[1].u[1] = *(const uint4*)&sP[wid][ln][32 + kb + 8];
    }

    // ---- O += P * V ----
#pragma unroll
    for (int nt = 0; nt < 4; ++nt) {
      FragBF vf0, vf1;
      const int dcol = nt * 16 + ln;     // dh column
      const int kb2 = halo * 16;         // key chunk
      vf0.u[0] = *(const uint4*)&sVT[dcol][kb2];
      vf0.u[1] = *(const uint4*)&sVT[dcol][kb2 + 8];
      vf1.u[0] = *(const uint4*)&sVT[dcol][32 + kb2];
      vf1.u[1] = *(const uint4*)&sVT[dcol][32 + kb2 + 8];
      O[nt] = wmma_bf16(pf[0].v, vf0.v, O[nt]);
      O[nt] = wmma_bf16(pf[1].v, vf1.v, O[nt]);
    }
    __syncthreads();
  }

  // ---- normalize + store context (bf16, [B,SQ,H*DH]) ----
#pragma unroll
  for (int nt = 0; nt < 4; ++nt)
#pragma unroll
    for (int r = 0; r < 8; ++r) {
      const int qidx = q0 + wid * 16 + r + 8 * halo;
      const float o = O[nt][r] / lrun[r];
      Ctx[((size_t)(b * SQn + qidx)) * (Hh * DHd) + h * DHd + nt * 16 + ln] = f2bf(o);
    }
}

// ---------------- host orchestration ----------------
extern "C" void kernel_launch(void* const* d_in, const int* in_sizes, int n_in,
                              void* d_out, int out_size, void* d_ws, size_t ws_size,
                              hipStream_t stream) {
  (void)in_sizes; (void)n_in; (void)out_size; (void)ws_size;
  const float* x   = (const float*)d_in[0];
  const float* enc = (const float*)d_in[1];
  const float* Wq  = (const float*)d_in[2];
  const float* Wk  = (const float*)d_in[3];
  const float* Wv  = (const float*)d_in[4];
  const float* Wo  = (const float*)d_in[5];
  const float* ln0 = (const float*)d_in[6];
  const float* ln1 = (const float*)d_in[7];
  const float* wi0 = (const float*)d_in[8];
  const float* wi1 = (const float*)d_in[9];
  const float* wof = (const float*)d_in[10];
  const float* rb  = (const float*)d_in[11];
  const int* buckets = (const int*)d_in[12];
  float* out = (float*)d_out;

  char* ws = (char*)d_ws;
  const size_t MB = 1024ull * 1024ull;
  unsigned short* enc_bf = (unsigned short*)(ws + 0 * MB);   // 8 MB
  unsigned short* h_bf   = (unsigned short*)(ws + 8 * MB);   // 8 MB (h, reused as h2)
  unsigned short* wq_bf  = (unsigned short*)(ws + 16 * MB);  // 2 MB  [N][K]
  unsigned short* wk_bf  = (unsigned short*)(ws + 18 * MB);
  unsigned short* wv_bf  = (unsigned short*)(ws + 20 * MB);
  unsigned short* wo_bf  = (unsigned short*)(ws + 22 * MB);
  unsigned short* wi0_bf = (unsigned short*)(ws + 24 * MB);  // 8 MB [FF][D]
  unsigned short* wi1_bf = (unsigned short*)(ws + 32 * MB);  // 8 MB
  unsigned short* wof_bf = (unsigned short*)(ws + 40 * MB);  // 8 MB [D][FF]
  unsigned short* q_bf   = (unsigned short*)(ws + 48 * MB);  // 8 MB
  unsigned short* k_bf   = (unsigned short*)(ws + 56 * MB);
  unsigned short* vT_bf  = (unsigned short*)(ws + 64 * MB);  // 8 MB [b][h][d][key]
  unsigned short* ctx_bf = (unsigned short*)(ws + 72 * MB);  // 8 MB
  unsigned short* g_bf   = (unsigned short*)(ws + 80 * MB);  // 32 MB (end 112 MB)

  const int TPB = 256;
  {
    int n = Bn * SKn * Dm;
    cvt_f32_bf16_kernel<<<(n + TPB * 4 - 1) / (TPB * 4), TPB, 0, stream>>>(enc, enc_bf, n);
  }
  auto cvtT = [&](const float* s, unsigned short* d, int K, int N) {
    dim3 g(N / 32, K / 32);
    cvt_transpose_bf16_kernel<<<g, 256, 0, stream>>>(s, d, K, N);
  };
  cvtT(Wq, wq_bf, Dm, Hh * DHd);
  cvtT(Wk, wk_bf, Dm, Hh * DHd);
  cvtT(Wv, wv_bf, Dm, Hh * DHd);
  cvtT(Wo, wo_bf, Hh * DHd, Dm);
  cvtT(wi0, wi0_bf, Dm, FFn);
  cvtT(wi1, wi1_bf, Dm, FFn);
  cvtT(wof, wof_bf, FFn, Dm);

  // --- cross-attention sublayer ---
  rmsnorm_bf16_kernel<<<Bn * SQn, 256, 0, stream>>>(x, ln0, h_bf);

  dim3 gP(Dm / BN, (Bn * SQn) / BM);           // (16, 32)
  gemm_bf16_kernel<2><<<gP, 256, 0, stream>>>(h_bf, wq_bf, nullptr, nullptr, nullptr,
                                              q_bf, Bn * SQn, Hh * DHd, Dm);
  gemm_bf16_kernel<2><<<gP, 256, 0, stream>>>(enc_bf, wk_bf, nullptr, nullptr, nullptr,
                                              k_bf, Bn * SKn, Hh * DHd, Dm);
  gemm_bf16_kernel<4><<<gP, 256, 0, stream>>>(enc_bf, wv_bf, nullptr, nullptr, nullptr,
                                              vT_bf, Bn * SKn, Hh * DHd, Dm);

  dim3 gA(SQn / 64, Hh, Bn);
  flash_attn_kernel<<<gA, 128, 0, stream>>>(q_bf, k_bf, vT_bf, rb, buckets, ctx_bf);

  // x = x + ctx @ Wo   (writes d_out)
  gemm_bf16_kernel<1><<<gP, 256, 0, stream>>>(ctx_bf, wo_bf, nullptr, x, out, nullptr,
                                              Bn * SQn, Dm, Hh * DHd);

  // --- gated GELU FFN sublayer ---
  rmsnorm_bf16_kernel<<<Bn * SQn, 256, 0, stream>>>(out, ln1, h_bf);

  dim3 gF(FFn / BN, (Bn * SQn) / BM);          // (64, 32)
  gemm_bf16_kernel<3><<<gF, 256, 0, stream>>>(h_bf, wi0_bf, wi1_bf, nullptr, nullptr,
                                              g_bf, Bn * SQn, FFn, Dm);

  // out = out + g @ wo_ffn
  gemm_bf16_kernel<1><<<gP, 256, 0, stream>>>(g_bf, wof_bf, nullptr, out, out, nullptr,
                                              Bn * SQn, Dm, FFn);
}
